// PredictTagsLayer_17875653886244
// MI455X (gfx1250) — matrix operand order repaired and checked by reading, not compiled, not run on hardware
//
#include <hip/hip_runtime.h>

// ---------------------------------------------------------------------------
// MI455X (gfx1250): fused WMMA GEMM (split-f16 compensation) + streaming top-k.
// scores[u,t] = <U[u,:], T[t,:]>, output = top-20 tag indices per user (int32).
// Pass 0 splits T into f16 hi + f16 residual lo in d_ws; pass 1 streams it.
// Hot loop (PRE): 4x global_load_b128 + 3x v_wmma per k-tile, A in registers.
// ---------------------------------------------------------------------------

typedef __attribute__((ext_vector_type(16))) _Float16 v16h;
typedef __attribute__((ext_vector_type(8)))  _Float16 v8h;
typedef __attribute__((ext_vector_type(8)))  float    v8f;

#define N_USERS 2048
#define N_TAGS  100000
#define DIM     256
#define TOPK    20
#define WAVES   8
#define NTILES  (N_TAGS / 16)            // 6250
#define NEG_INF (-3.4e38f)
#define APAD    8                        // LDS row pad (halves) -> bank-conflict free
#define AST     (DIM + APAD)             // 264 halves = 528 B row stride

__device__ __forceinline__ int perm_i(int v, int srcLane) {
  return __builtin_amdgcn_ds_bpermute(srcLane << 2, v);
}
__device__ __forceinline__ float perm_f(float v, int srcLane) {
  return __builtin_bit_cast(float,
      __builtin_amdgcn_ds_bpermute(srcLane << 2, __builtin_bit_cast(int, v)));
}
__device__ __forceinline__ v16h cat(v8h a, v8h b) {
  return __builtin_shufflevector(a, b, 0, 1, 2, 3, 4, 5, 6, 7,
                                 8, 9, 10, 11, 12, 13, 14, 15);
}
// 8 fp32 -> f16 hi + f16 residual lo.
__device__ __forceinline__ void cvt8(const float4 a, const float4 b,
                                     v8h& h, v8h& l) {
  _Float16 h0 = (_Float16)a.x, h1 = (_Float16)a.y;
  _Float16 h2 = (_Float16)a.z, h3 = (_Float16)a.w;
  _Float16 h4 = (_Float16)b.x, h5 = (_Float16)b.y;
  _Float16 h6 = (_Float16)b.z, h7 = (_Float16)b.w;
  h = (v8h){h0, h1, h2, h3, h4, h5, h6, h7};
  l = (v8h){(_Float16)(a.x - (float)h0), (_Float16)(a.y - (float)h1),
            (_Float16)(a.z - (float)h2), (_Float16)(a.w - (float)h3),
            (_Float16)(b.x - (float)h4), (_Float16)(b.y - (float)h5),
            (_Float16)(b.z - (float)h6), (_Float16)(b.w - (float)h7)};
}

// ---- Pass 0: split fp32 tensor into f16 hi / f16 lo tables (8 elems/thread).
__global__ __launch_bounds__(256) void split_f16_kernel(
    const float* __restrict__ src, _Float16* __restrict__ hi,
    _Float16* __restrict__ lo, int n8) {
  const int g = blockIdx.x * 256 + threadIdx.x;
  if (g >= n8) return;
  const float4 a = *(const float4*)(src + g * 8);
  const float4 b = *(const float4*)(src + g * 8 + 4);
  v8h h, l;
  cvt8(a, b, h, l);
  *(v8h*)(hi + g * 8) = h;
  *(v8h*)(lo + g * 8) = l;
}

// ---- Pass 1: fused GEMM + top-k. PRE=true reads preconverted Thi/Tlo.
template <bool PRE>
__global__ __launch_bounds__(256) void predict_tags_topk(
    const float* __restrict__ U, const float* __restrict__ Tf,
    const _Float16* __restrict__ Thi, const _Float16* __restrict__ Tlo,
    int* __restrict__ out) {
  __shared__ float    s_sc[WAVES][16][TOPK];
  __shared__ int      s_id[WAVES][16][TOPK];
  __shared__ _Float16 sAhi[16][AST];     // A staging, padded: bank = 4*lane
  __shared__ _Float16 sAlo[16][AST];

  const int tid  = threadIdx.x;
  const int lane = tid & 31;
  const int wv   = tid >> 5;
  const int half = lane >> 4;
  const int l15  = lane & 15;

  // Init this wave's private top-k lists.
  for (int i = lane; i < 16 * TOPK; i += 32) {
    s_sc[wv][i / TOPK][i % TOPK] = NEG_INF;
    s_id[wv][i / TOPK][i % TOPK] = 0;
  }

  // Convert A tile (16 users x 256) to split-f16 in LDS; 16 elems/thread.
  {
    const int base = tid * 16;           // 0..4095
    const int u = base >> 8, k = base & 255;
    const float* up = U + (blockIdx.x * 16 + u) * DIM + k;
    v8h h0, l0, h1, l1;
    cvt8(*(const float4*)(up), *(const float4*)(up + 4), h0, l0);
    cvt8(*(const float4*)(up + 8), *(const float4*)(up + 12), h1, l1);
    *(v8h*)&sAhi[u][k]     = h0;  *(v8h*)&sAlo[u][k]     = l0;
    *(v8h*)&sAhi[u][k + 8] = h1;  *(v8h*)&sAlo[u][k + 8] = l1;
  }
  __syncthreads();

  // PRE: A operands are tile-invariant -> hoist all 8 k-tiles (hi+lo) into
  // 128 VGPRs once; hot loop then touches no LDS at all.
  v16h Ah[8], Al[8];
  if constexpr (PRE) {
#pragma unroll
    for (int kt = 0; kt < 8; ++kt) {
      const int k0 = kt * 32 + half * 16;
      Ah[kt] = cat(*(const v8h*)&sAhi[l15][k0],
                   *(const v8h*)&sAhi[l15][k0 + 8]);
      Al[kt] = cat(*(const v8h*)&sAlo[l15][k0],
                   *(const v8h*)&sAlo[l15][k0 + 8]);
    }
  }

  // Stream all tag tiles; wave wv takes tiles wv, wv+8, ...
  for (int tile = wv; tile < NTILES; tile += WAVES) {
    const int tag = tile * 16 + l15;     // B column this lane owns
    {
      int pt = tile + WAVES; if (pt >= NTILES) pt = NTILES - 1;
      const int pofs = (pt * 16 + l15) * DIM;
      if (PRE) { __builtin_prefetch(Thi + pofs, 0, 1);
                 __builtin_prefetch(Tlo + pofs, 0, 1); }
      else     { __builtin_prefetch(Tf + pofs, 0, 1); }
    }
    v8f acc = {0.f, 0.f, 0.f, 0.f, 0.f, 0.f, 0.f, 0.f};
#pragma unroll
    for (int kt = 0; kt < 8; ++kt) {
      // B operand: 16 contiguous K per lane (lanes 0-15: K 0..15; 16-31: 16..31)
      const int kb = kt * 32 + half * 16;
      v16h Bh, Bl, Ahk, Alk;
      if constexpr (PRE) {
        const _Float16* th = Thi + tag * DIM + kb;
        const _Float16* tl = Tlo + tag * DIM + kb;
        Bh = cat(*(const v8h*)(th), *(const v8h*)(th + 8));
        Bl = cat(*(const v8h*)(tl), *(const v8h*)(tl + 8));
        Ahk = Ah[kt]; Alk = Al[kt];
      } else {
        const float* tp = Tf + tag * DIM + kb;
        v8h h0, l0, h1, l1;
        cvt8(*(const float4*)(tp), *(const float4*)(tp + 4), h0, l0);
        cvt8(*(const float4*)(tp + 8), *(const float4*)(tp + 12), h1, l1);
        Bh = cat(h0, h1); Bl = cat(l0, l1);
        const int k0 = kt * 32 + half * 16;  // A from LDS (keeps VGPRs low)
        Ahk = cat(*(const v8h*)&sAhi[l15][k0],
                  *(const v8h*)&sAhi[l15][k0 + 8]);
        Alk = cat(*(const v8h*)&sAlo[l15][k0],
                  *(const v8h*)&sAlo[l15][k0 + 8]);
      }
      // Split-f16: hi*hi + hi*lo + lo*hi in one f32 accumulator.
      acc = __builtin_amdgcn_wmma_f32_16x16x32_f16(false, Ahk, false, Bh,
                                                   (short)0, acc, false, false);
      acc = __builtin_amdgcn_wmma_f32_16x16x32_f16(false, Ahk, false, Bl,
                                                   (short)0, acc, false, false);
      acc = __builtin_amdgcn_wmma_f32_16x16x32_f16(false, Alk, false, Bh,
                                                   (short)0, acc, false, false);
    }

    // Top-k update. C layout: lane l, VGPR r -> C[M = r + 8*(l>>4)][N = l&15].
#pragma unroll
    for (int r = 0; r < 8; ++r) {
      const int   u = r + 8 * half;
      const float s = acc[r];
      const bool cand = s > s_sc[wv][u][TOPK - 1];     // threshold gate
      unsigned int m = __builtin_amdgcn_ballot_w32(cand);
      while (m) {                                      // uniform loop
        const int src = __builtin_ctz(m);
        m &= m - 1;
        const float cs = perm_f(s, src);
        const int   ct = tile * 16 + (src & 15);
        if (lane == 0) {                               // serial sorted insert
          const int cu = r + 8 * (src >> 4);
          float* sl = &s_sc[wv][cu][0];
          int*   il = &s_id[wv][cu][0];
          if (cs > sl[TOPK - 1]) {
            int p = TOPK - 1;
            while (p > 0 && sl[p - 1] < cs) {
              sl[p] = sl[p - 1]; il[p] = il[p - 1]; --p;
            }
            sl[p] = cs; il[p] = ct;
          }
        }
      }
    }
  }

  __syncthreads();

  // Merge the 8 wave lists per user; wave wv handles users 2wv, 2wv+1.
  for (int uu = 0; uu < 2; ++uu) {
    const int u = 2 * wv + uu;
    float cs[5]; int ct[5];                            // 160 candidates/32 lanes
#pragma unroll
    for (int j = 0; j < 5; ++j) {
      const int c = lane + 32 * j;
      cs[j] = s_sc[c / TOPK][u][c % TOPK];
      ct[j] = s_id[c / TOPK][u][c % TOPK];
    }
    for (int oj = 0; oj < TOPK; ++oj) {
      float best = cs[0]; int bslot = 0;
#pragma unroll
      for (int j = 1; j < 5; ++j)
        if (cs[j] > best) { best = cs[j]; bslot = j; }
      int btag = ct[0];                                // ct[bslot] w/o dyn index
#pragma unroll
      for (int j = 1; j < 5; ++j) if (bslot == j) btag = ct[j];
      // Wave argmax butterfly (value desc, lane asc tie-break).
      float rb = best; int rl = lane;
#pragma unroll
      for (int off = 16; off > 0; off >>= 1) {
        const float ob = perm_f(rb, lane ^ off);
        const int   ol = perm_i(rl, lane ^ off);
        if (ob > rb || (ob == rb && ol < rl)) { rb = ob; rl = ol; }
      }
      if (lane == rl) out[u * TOPK + oj] = btag;
#pragma unroll
      for (int j = 0; j < 5; ++j)                      // kill winner slot
        if (lane == rl && bslot == j) cs[j] = NEG_INF;
    }
  }
}

extern "C" void kernel_launch(void* const* d_in, const int* in_sizes, int n_in,
                              void* d_out, int out_size, void* d_ws,
                              size_t ws_size, hipStream_t stream) {
  (void)in_sizes; (void)n_in; (void)out_size;
  const float* U  = (const float*)d_in[0];   // [2048, 256]
  const float* Tf = (const float*)d_in[1];   // [100000, 256]
  int* out = (int*)d_out;                    // [2048, 20] int32 indices

  const size_t t_elems = (size_t)N_TAGS * DIM;            // 25.6M
  const size_t need    = 2 * t_elems * sizeof(_Float16);  // 102.4 MB
  dim3 grid(N_USERS / 16), block(32 * WAVES);

  if (ws_size >= need) {
    _Float16* Thi = (_Float16*)d_ws;
    _Float16* Tlo = Thi + t_elems;
    const int n8 = (int)(t_elems / 8);                    // 3.2M groups
    split_f16_kernel<<<dim3((n8 + 255) / 256), dim3(256), 0, stream>>>(
        Tf, Thi, Tlo, n8);
    predict_tags_topk<true><<<grid, block, 0, stream>>>(U, Tf, Thi, Tlo, out);
  } else {
    predict_tags_topk<false><<<grid, block, 0, stream>>>(U, Tf, nullptr,
                                                         nullptr, out);
  }
}
